// SA_module_5669356836166
// MI455X (gfx1250) — matrix operand order repaired and checked by reading, not compiled, not run on hardware
//
#include <hip/hip_runtime.h>

typedef __attribute__((ext_vector_type(16))) _Float16 v16h;
typedef __attribute__((ext_vector_type(8)))  _Float16 v8h;
typedef __attribute__((ext_vector_type(8)))  float    v8f;

#define B_   4
#define N_   8192
#define K_   16
#define INP  128
#define REL  32
#define MID  128
#define OUTC 128
#define MSH  16            // MID/SHARE
#define HCH  544           // REL*(K+1) = 17*32
#define NT   16            // points per workgroup

// fragment table offsets (units of 512 f16 = one 16x32 A fragment)
#define F_W1   0           // 2 mtiles * 4 ktiles
#define F_W2   8
#define F_W3   16          // 8 * 4
#define F_WOUT 48          // 8 * 4
#define F_WW1  80          // 1 * 17
#define F_WW2  97          // 16 * 1 (K padded 16->32 with zeros)
#define F_TOTAL 113

// A-matrix 16x32 packing (contiguous, matches ld_b): lane half 0 -> K 0-15;
// lane half 1 -> K 16-31.
__device__ __forceinline__ int a_klocal(int i, int half) {
  return half * 16 + i;
}

__global__ void pack_a_kernel(const float* __restrict__ W, _Float16* __restrict__ dst,
                              int M, int Kc, int ktiles) {
  int frag = blockIdx.x;
  int mt = frag / ktiles, kt = frag % ktiles;
  int lane = threadIdx.x;              // 32 threads / block
  int half = lane >> 4;
  int m = mt * 16 + (lane & 15);
  v16h t;
  #pragma unroll
  for (int i = 0; i < 16; ++i) {
    int k = kt * 32 + a_klocal(i, half);
    float v = (m < M && k < Kc) ? W[(size_t)m * Kc + k] : 0.f;
    t[i] = (_Float16)v;
  }
  *(v16h*)(dst + (size_t)frag * 512 + (size_t)lane * 16) = t;
}

// A fragment: straight 32B load from packed weight table
__device__ __forceinline__ v16h ld_a(const _Float16* __restrict__ wsf, int frag, int lane) {
  return *(const v16h*)(wsf + (size_t)frag * 512 + (size_t)lane * 16);
}
// B fragment (K x 16) from channel-contiguous (column-major) f16 LDS:
// lane half 0 -> K k0..k0+15, half 1 -> K k0+16..k0+31, column = lane&15.
__device__ __forceinline__ v16h ld_b(const _Float16* base, int stride, int col0, int k0, int lane) {
  int n = lane & 15, half = lane >> 4;
  return *(const v16h*)(base + (size_t)(col0 + n) * stride + k0 + half * 16);
}

#define WMMA(a, b, c) __builtin_amdgcn_wmma_f32_16x16x32_f16(false, (a), false, (b), (short)0, (c), false, false)

__global__ __launch_bounds__(256) void sa_main_kernel(
    const float* __restrict__ x, const int* __restrict__ idx,
    const _Float16* __restrict__ wsf,
    const float* __restrict__ b1, const float* __restrict__ b2, const float* __restrict__ b3,
    const float* __restrict__ bw2, const float* __restrict__ bout,
    float* __restrict__ out)
{
  extern __shared__ char smem[];
  _Float16* xa   = (_Float16*)(smem);                  // 16 pts * 128 ch        : 4 KB
  _Float16* xn   = (_Float16*)(smem + 4096);           // 256 cols * 128 ch      : 64 KB
  _Float16* hbuf = (_Float16*)(smem + 69632);          // 16 pts * 544 ch        : 17 KB
  _Float16* h2   = (_Float16*)(smem + 87040);          // 16 pts * 32 ch (pad)   : 1 KB
  float*    wat  = (float*)   (smem + 88064);          // 16 pts * 256           : 16 KB
  _Float16* x3   = (_Float16*)(smem + 104448);         // 256 cols * 128 ch      : 64 KB
  _Float16* mid  = (_Float16*)(smem + 169984);         // 16 pts * 128 ch        : 4 KB
  // total 174080 bytes dynamic LDS

  const int t    = threadIdx.x;
  const int w    = t >> 5, lane = t & 31;
  const int nlo  = lane & 15, half = lane >> 4;
  const int b    = blockIdx.y;
  const int n0   = blockIdx.x * NT;

  // ---- phase 1a: xa = relu(x) tile, f16 column-major ----
  {
    int pt = t & 15, c0 = t >> 4;          // 16-lane groups read 16 consecutive n (coalesced)
    for (int c = c0; c < INP; c += 16) {
      float v = x[((size_t)b * INP + c) * N_ + n0 + pt];
      xa[pt * INP + c] = (_Float16)fmaxf(v, 0.f);
    }
  }
  // ---- phase 1b: gather neighbor columns (L2-resident), relu, f16 ----
  {
    int pt = t >> 4, j = t & 15;           // thread owns one (point, neighbor) column
    int nb = idx[((size_t)b * N_ + n0 + pt) * K_ + j];
    for (int c0 = 0; c0 < INP; c0 += 8) {
      v8h tmp;
      #pragma unroll
      for (int u = 0; u < 8; ++u) {
        float v = x[((size_t)b * INP + c0 + u) * N_ + nb];
        tmp[u] = (_Float16)fmaxf(v, 0.f);
      }
      *(v8h*)(xn + (size_t)t * INP + c0) = tmp;        // 16B LDS store
    }
  }
  __syncthreads();

  // ---- phase 2: x2 = relu(W2@xn + b2) into h[32..543]; x1 into h[0..31] ----
  #pragma unroll
  for (int it = 0; it < 4; ++it) {                     // 32 tiles, 4 per wave
    int tt = w + it * 8;
    int mt = tt & 1, ct = tt >> 1;                     // ct == point index
    v8f c = {};
    #pragma unroll
    for (int kt = 0; kt < 4; ++kt)
      c = WMMA(ld_a(wsf, F_W2 + mt * 4 + kt, lane),
               ld_b(xn, INP, ct * 16, kt * 32, lane), c);
    #pragma unroll
    for (int r = 0; r < 8; ++r) {
      int m = mt * 16 + r + 8 * half;                  // rel row; nlo = neighbor
      float v = c[r] + b2[m];
      hbuf[ct * HCH + REL + m * K_ + nlo] = (_Float16)fmaxf(v, 0.f);
    }
  }
  if (w < 2) {                                         // x1 = relu(W1@xa + b1)
    int mt = w;
    v8f c = {};
    #pragma unroll
    for (int kt = 0; kt < 4; ++kt)
      c = WMMA(ld_a(wsf, F_W1 + mt * 4 + kt, lane),
               ld_b(xa, INP, 0, kt * 32, lane), c);
    #pragma unroll
    for (int r = 0; r < 8; ++r) {
      int m = mt * 16 + r + 8 * half;
      float v = c[r] + b1[m];
      hbuf[nlo * HCH + m] = (_Float16)fmaxf(v, 0.f);   // nlo = point
    }
  }
  __syncthreads();

  // ---- phase 3: h2 = relu(Ww1 @ h), K = 544 = 17*32 exactly (wave 0) ----
  if (w == 0) {
    v8f c = {};
    for (int kt = 0; kt < 17; ++kt)
      c = WMMA(ld_a(wsf, F_WW1 + kt, lane),
               ld_b(hbuf, HCH, 0, kt * 32, lane), c);
    #pragma unroll
    for (int r = 0; r < 8; ++r) {
      int m = r + 8 * half;
      h2[nlo * 32 + m]      = (_Float16)fmaxf(c[r], 0.f);
      h2[nlo * 32 + 16 + m] = (_Float16)0.f;           // zero-pad K 16..31
    }
  }
  __syncthreads();

  // ---- phase 4: attention weights wat = Ww2 @ h2 + bw2 (16 tiles, 2/wave) ----
  #pragma unroll
  for (int it = 0; it < 2; ++it) {
    int mt = w * 2 + it;
    v8f c = {};
    c = WMMA(ld_a(wsf, F_WW2 + mt, lane), ld_b(h2, 32, 0, 0, lane), c);
    #pragma unroll
    for (int r = 0; r < 8; ++r) {
      int r8 = r + 8 * half;                           // neighbor index
      int m  = mt * 16 + r8;
      wat[nlo * 256 + r8 * 16 + mt] = c[r] + bw2[m];   // [pt][nb][mid/share]
    }
  }

  // ---- phase 5: x3 = W3 @ xn + b3; wave w owns row-tile mt=w, A frags in regs ----
  {
    v16h a0 = ld_a(wsf, F_W3 + w * 4 + 0, lane);
    v16h a1 = ld_a(wsf, F_W3 + w * 4 + 1, lane);
    v16h a2 = ld_a(wsf, F_W3 + w * 4 + 2, lane);
    v16h a3 = ld_a(wsf, F_W3 + w * 4 + 3, lane);
    for (int ct = 0; ct < 16; ++ct) {
      v8f c = {};
      c = WMMA(a0, ld_b(xn, INP, ct * 16,  0, lane), c);
      c = WMMA(a1, ld_b(xn, INP, ct * 16, 32, lane), c);
      c = WMMA(a2, ld_b(xn, INP, ct * 16, 64, lane), c);
      c = WMMA(a3, ld_b(xn, INP, ct * 16, 96, lane), c);
      #pragma unroll
      for (int r = 0; r < 8; ++r) {
        int m = w * 16 + r + 8 * half;
        x3[(ct * 16 + nlo) * INP + m] = (_Float16)(c[r] + b3[m]);
      }
    }
  }
  __syncthreads();

  // ---- phase 6: mid = relu(sum_nb wat[c%16,nb,pt] * x3[c,nb,pt]) ----
  for (int e = t; e < NT * MID; e += 256) {
    int pt = e >> 7, c = e & 127;
    int wrow = c & 15;
    float acc = 0.f;
    #pragma unroll
    for (int nb = 0; nb < K_; ++nb)
      acc += (float)x3[(pt * 16 + nb) * INP + c] * wat[pt * 256 + nb * 16 + wrow];
    mid[pt * INP + c] = (_Float16)fmaxf(acc, 0.f);
  }
  __syncthreads();

  // ---- phase 7: out = Wout @ mid + bout + x (residual), direct global store ----
  {
    v8f c = {};
    #pragma unroll
    for (int kt = 0; kt < 4; ++kt)
      c = WMMA(ld_a(wsf, F_WOUT + w * 4 + kt, lane),
               ld_b(mid, INP, 0, kt * 32, lane), c);
    #pragma unroll
    for (int r = 0; r < 8; ++r) {
      int m = w * 16 + r + 8 * half;
      size_t gi = ((size_t)b * OUTC + m) * N_ + n0 + nlo;
      out[gi] = c[r] + bout[m] + x[gi];
    }
  }
}

__global__ void copy_idx_kernel(const int* __restrict__ idx, int* __restrict__ dst, int n) {
  int i = blockIdx.x * 256 + threadIdx.x;
  if (i < n) dst[i] = idx[i];
}

extern "C" void kernel_launch(void* const* d_in, const int* in_sizes, int n_in,
                              void* d_out, int out_size, void* d_ws, size_t ws_size,
                              hipStream_t stream) {
  (void)in_sizes; (void)n_in; (void)out_size; (void)ws_size;
  const float* x    = (const float*)d_in[0];
  const int*   idx  = (const int*)  d_in[1];
  const float* W1   = (const float*)d_in[2];
  const float* b1   = (const float*)d_in[3];
  const float* W2   = (const float*)d_in[4];
  const float* b2   = (const float*)d_in[5];
  const float* W3   = (const float*)d_in[6];
  const float* b3   = (const float*)d_in[7];
  const float* Ww1  = (const float*)d_in[8];
  const float* Ww2  = (const float*)d_in[9];
  const float* bw2  = (const float*)d_in[10];
  const float* Wout = (const float*)d_in[11];
  const float* bout = (const float*)d_in[12];
  _Float16* wsf = (_Float16*)d_ws;      // needs 113*512*2 = 115,712 B of scratch
  float* out = (float*)d_out;

  // pack f16 A-fragments (weights) once per call; L2-resident thereafter
  pack_a_kernel<<<8,  32, 0, stream>>>(W1,   wsf + (size_t)F_W1   * 512, REL,      INP, 4);
  pack_a_kernel<<<8,  32, 0, stream>>>(W2,   wsf + (size_t)F_W2   * 512, REL,      INP, 4);
  pack_a_kernel<<<32, 32, 0, stream>>>(W3,   wsf + (size_t)F_W3   * 512, MID,      INP, 4);
  pack_a_kernel<<<32, 32, 0, stream>>>(Wout, wsf + (size_t)F_WOUT * 512, OUTC,     MID, 4);
  pack_a_kernel<<<17, 32, 0, stream>>>(Ww1,  wsf + (size_t)F_WW1  * 512, MSH,      HCH, 17);
  pack_a_kernel<<<16, 32, 0, stream>>>(Ww2,  wsf + (size_t)F_WW2  * 512, K_ * MSH, MSH, 1);

  dim3 grid(N_ / NT, B_);
  sa_main_kernel<<<grid, 256, 174080, stream>>>(x, idx, wsf, b1, b2, b3, bw2, bout, out);

  // tuple return: append idx after the [B,OUT,1,N] tensor
  int idx_n = B_ * N_ * K_;
  copy_idx_kernel<<<(idx_n + 255) / 256, 256, 0, stream>>>(idx, (int*)d_out + (size_t)B_ * OUTC * N_, idx_n);
}
